// AttentionSAT_80247168958870
// MI455X (gfx1250) — matrix operand order repaired and checked by reading, not compiled, not run on hardware
//
#include <hip/hip_runtime.h>
#include <hip/hip_bf16.h>
#include <math.h>

// ---- problem constants (match reference) ----
#define NVARS 10000
#define NLITS 20000
#define NCLS  42000
#define EDG   126000
#define FDIM  128
#define QD    64
#define UNITD 224
#define ROUNDS 4

typedef __attribute__((ext_vector_type(16))) _Float16 v16h;
typedef __attribute__((ext_vector_type(8)))  _Float16 v8h;
typedef __attribute__((ext_vector_type(8)))  float    v8f;

__device__ __forceinline__ float softplusf(float x) {
  if (x > 20.f)  return x;
  if (x < -20.f) return __expf(x);
  return log1pf(__expf(x));
}

// ============================================================
// Generic WMMA GEMM:  C[M,N] = A[M,Kp] * B[Kp,N]
//   A: row-major f16 (Kp multiple of 32, zero-padded)
//   Bt: weights pre-transposed col-major f16, Bt[n*Kp + k]
//   grid.x = M/16, blockDim = 32*(N/16)  (N = 64 or 128)
//   mode 0: linear -> outF32 / outF16 (nullable)
//   mode 1: relu -> row LayerNorm -> outF32 / outF16 (N must be 128)
// ============================================================
__global__ __launch_bounds__(256) void gemm_wmma_kernel(
    const _Float16* __restrict__ A, const _Float16* __restrict__ Bt,
    float* __restrict__ outF32, _Float16* __restrict__ outF16,
    int Kp, int N, int mode)
{
  __shared__ float lds[16 * 128];
  const int wave = threadIdx.x >> 5;
  const int lane = threadIdx.x & 31;
  const int m    = lane & 15;   // row within tile (A) / col within tile (B,C)
  const int h    = lane >> 4;   // half-wave select
  const int tileRow = blockIdx.x;

  v8f acc = {};
  const _Float16* aRow = A  + (size_t)(tileRow * 16 + m) * Kp;
  const _Float16* bCol = Bt + (size_t)(wave * 16 + m) * Kp;
  for (int kk = 0; kk < Kp; kk += 32) {
    // A fragment: lane(m,h): elems j = K(kk+16h+j)  (contiguous)
    v8h a0 = *(const v8h*)(aRow + kk + 16 * h);
    v8h a1 = *(const v8h*)(aRow + kk + 16 * h + 8);
    v16h af;
#pragma unroll
    for (int i = 0; i < 8; ++i) { af[i] = a0[i]; af[i + 8] = a1[i]; }
    // B fragment: lane(n=m,h): elems j = B[kk+16h+j][n]  (contiguous in Bt)
    v16h bf = *(const v16h*)(bCol + kk + 16 * h);
    acc = __builtin_amdgcn_wmma_f32_16x16x32_f16(false, af, false, bf,
                                                 (short)0, acc, false, false);
  }
  // C layout: VGPR j -> row (8h + j), col n = m
#pragma unroll
  for (int j = 0; j < 8; ++j)
    lds[(8 * h + j) * N + wave * 16 + m] = acc[j];
  __syncthreads();

  if (mode == 0) {
    for (int idx = threadIdx.x; idx < 16 * N; idx += blockDim.x) {
      int r = idx / N, c = idx - r * N;
      float v = lds[r * N + c];
      size_t o = (size_t)(tileRow * 16 + r) * N + c;
      if (outF32) outF32[o] = v;
      if (outF16) outF16[o] = (_Float16)v;
    }
  } else {
    // relu -> per-row LayerNorm; N == 128, 8 waves, wave w owns rows 2w, 2w+1.
    // 32 lanes cover 128 cols (4 each); wave32 shuffle reductions for mu/var.
#pragma unroll
    for (int rr = 0; rr < 2; ++rr) {
      int r = wave * 2 + rr;
      float v[4];
      float mu = 0.f;
#pragma unroll
      for (int t = 0; t < 4; ++t) {
        float x = lds[r * 128 + lane + 32 * t];
        x = x > 0.f ? x : 0.f;
        v[t] = x; mu += x;
      }
#pragma unroll
      for (int off = 16; off > 0; off >>= 1) mu += __shfl_down(mu, off, 32);
      mu = __shfl(mu, 0, 32) * (1.f / 128.f);
      float s2 = 0.f;
#pragma unroll
      for (int t = 0; t < 4; ++t) { float d = v[t] - mu; s2 += d * d; }
#pragma unroll
      for (int off = 16; off > 0; off >>= 1) s2 += __shfl_down(s2, off, 32);
      float rstd = rsqrtf(__shfl(s2, 0, 32) * (1.f / 128.f) + 1e-6f);
#pragma unroll
      for (int t = 0; t < 4; ++t) {
        float x = (v[t] - mu) * rstd;
        size_t o = (size_t)(tileRow * 16 + r) * 128 + lane + 32 * t;
        if (outF32) outF32[o] = x;
        if (outF16) outF16[o] = (_Float16)x;
      }
    }
  }
}

// ---- small utility kernels ----
__global__ void zero_f32_kernel(float* p, int n) {
  int i = blockIdx.x * blockDim.x + threadIdx.x;
  if (i < n) p[i] = 0.f;
}
__global__ void zero_u32_kernel(unsigned* p, int n) {
  int i = blockIdx.x * blockDim.x + threadIdx.x;
  if (i < n) p[i] = 0u;
}
// W[K,N] f32 -> Wt[N,Kp] f16 (transposed, zero-padded K)
__global__ void wcvt_kernel(const float* __restrict__ W, _Float16* __restrict__ Wt,
                            int K, int N, int Kp) {
  int i = blockIdx.x * blockDim.x + threadIdx.x;
  if (i >= N * Kp) return;
  int n = i / Kp, k = i - n * Kp;
  Wt[i] = (k < K) ? (_Float16)W[(size_t)k * N + n] : (_Float16)0.f;
}
// l_output zero-state: (onehot(0) - 1/F) * sqrt(F)*0.25
__global__ void init_lout_kernel(float* l) {
  int i = blockIdx.x * blockDim.x + threadIdx.x;
  if (i >= NLITS * FDIM) return;
  int c = i & (FDIM - 1);
  const float S = 2.82842712474619f; // sqrt(128)*0.25
  l[i] = ((c == 0 ? 1.f : 0.f) - 1.f / (float)FDIM) * S;
}
// Xvq[v, 0:288] = [lout[v], noise[v], lout[v+NV], noise[v+NV], pad]
__global__ void pack_vq_kernel(const float* __restrict__ lout,
                               const float* __restrict__ noise,
                               _Float16* __restrict__ X) {
  int i = blockIdx.x * blockDim.x + threadIdx.x;
  if (i >= NVARS * 288) return;
  int v = i / 288, c = i - v * 288;
  float val = 0.f;
  if (c < 128)      val = lout[v * FDIM + c];
  else if (c < 132) val = noise[v * 4 + (c - 128)];
  else if (c < 260) val = lout[(v + NVARS) * FDIM + (c - 132)];
  else if (c < 264) val = noise[(v + NVARS) * 4 + (c - 260)];
  X[i] = (_Float16)val;
}
// per (clause,k): clauses_loss, analytic inner grad scatter, literals_loss scatter
__global__ void clause_kernel(const float* __restrict__ query,
                              const int* __restrict__ cvar, const int* __restrict__ csign,
                              float* __restrict__ cl_out, float* __restrict__ lg,
                              float* __restrict__ llo) {
  int i = blockIdx.x * blockDim.x + threadIdx.x;
  if (i >= NCLS * QD) return;
  int c = i >> 6, k = i & 63;
  int e0 = 3 * c;
  int var[3], sb[3]; float z[3], sg[3];
  float s = 0.f;
#pragma unroll
  for (int t = 0; t < 3; ++t) {
    var[t] = cvar[e0 + t]; sb[t] = csign[e0 + t];
    sg[t] = 1.f - 2.f * (float)sb[t];
    z[t] = query[var[t] * QD + k] * sg[t];
    s += softplusf(z[t]);
  }
  float cl = __expf(-s);
  cl_out[i] = cl;
#pragma unroll
  for (int t = 0; t < 3; ++t) {
    float sig = 1.f / (1.f + __expf(-z[t]));
    atomicAdd(&lg[var[t] * QD + k], -cl * sig * sg[t]);   // d sum(exp(-s))/dq
    int lit = var[t] + NVARS * sb[t];
    atomicAdd(&llo[lit * QD + k], cl);                    // literals_loss
  }
}
__global__ void litunit_kernel(const float* __restrict__ lout, const float* __restrict__ lg,
                               const float* __restrict__ llo,
                               float* __restrict__ uf, _Float16* __restrict__ uh) {
  int i = blockIdx.x * blockDim.x + threadIdx.x;
  if (i >= NLITS * UNITD) return;
  int lit = i / UNITD, j = i - lit * UNITD;
  float v;
  if (j < 128) v = lout[lit * FDIM + j];
  else if (j < 160) {
    int jj = j - 128;
    v = (lit < NVARS) ? lg[lit * QD + jj] : lg[(lit - NVARS) * QD + 32 + jj];
  } else v = llo[lit * QD + (j - 160)];
  uf[i] = v; uh[i] = (_Float16)v;
}
__global__ void clsunit_kernel(const float* __restrict__ lout, const float* __restrict__ lg,
                               const float* __restrict__ cl,
                               const int* __restrict__ cvar, const int* __restrict__ csign,
                               float* __restrict__ uf, _Float16* __restrict__ uh) {
  int i = blockIdx.x * blockDim.x + threadIdx.x;
  if (i >= NCLS * UNITD) return;
  int c = i / UNITD, j = i - c * UNITD;
  float v;
  if (j >= 160) v = cl[c * QD + (j - 160)];
  else {
    v = 0.f;
    int e0 = 3 * c;
#pragma unroll
    for (int t = 0; t < 3; ++t) {
      int lit = cvar[e0 + t] + NVARS * csign[e0 + t];
      if (j < 128) v += lout[lit * FDIM + j];
      else {
        int jj = j - 128;
        v += (lit < NVARS) ? lg[lit * QD + jj] : lg[(lit - NVARS) * QD + 32 + jj];
      }
    }
  }
  uf[i] = v; uh[i] = (_Float16)v;
}
// one wave per edge: score = tanh(qp[lit]+mp[c]) . att_v ; atomicMax key into smax[lit]
__global__ void score_kernel(const float* __restrict__ qp, const float* __restrict__ mp,
                             const float* __restrict__ attv,
                             const int* __restrict__ cvar, const int* __restrict__ csign,
                             float* __restrict__ score, unsigned* __restrict__ smax) {
  int e = blockIdx.x * (blockDim.x >> 5) + (threadIdx.x >> 5);
  int lane = threadIdx.x & 31;
  if (e >= EDG) return;
  int c = e / 3;
  int lit = cvar[e] + NVARS * csign[e];
  float s = 0.f;
  for (int k = lane; k < FDIM; k += 32)
    s += tanhf(qp[lit * FDIM + k] + mp[c * FDIM + k]) * attv[k];
  for (int off = 16; off > 0; off >>= 1) s += __shfl_down(s, off, 32);
  if (lane == 0) {
    score[e] = s;
    unsigned b = __float_as_uint(s);
    unsigned key = (b & 0x80000000u) ? ~b : (b | 0x80000000u);
    atomicMax(&smax[lit], key);
  }
}
// per (edge,j): num[lit] += e * clauses_unit[c]; den[lit] += e
__global__ void att_acc_kernel(const float* __restrict__ score, const unsigned* __restrict__ smax,
                               const int* __restrict__ cvar, const int* __restrict__ csign,
                               const float* __restrict__ clsu,
                               float* __restrict__ num, float* __restrict__ den) {
  int i = blockIdx.x * blockDim.x + threadIdx.x;
  if (i >= EDG * UNITD) return;
  int e = i / UNITD, j = i - e * UNITD;
  int c = e / 3;
  int lit = cvar[e] + NVARS * csign[e];
  unsigned key = smax[lit];
  float mx = (key & 0x80000000u) ? __uint_as_float(key & 0x7FFFFFFFu) : __uint_as_float(~key);
  float ev = __expf(score[e] - mx);
  if (j == 0) atomicAdd(&den[lit], ev);
  atomicAdd(&num[lit * UNITD + j], ev * clsu[c * UNITD + j]);
}
// Xlm[lit] = [literals_unit(224) | flipped new_literals(224)]
__global__ void pack_lm_kernel(const float* __restrict__ litu, const float* __restrict__ num,
                               const float* __restrict__ den, _Float16* __restrict__ X) {
  int i = blockIdx.x * blockDim.x + threadIdx.x;
  if (i >= NLITS * 448) return;
  int lit = i / 448, j = i - lit * 448;
  float v;
  if (j < UNITD) v = litu[lit * UNITD + j];
  else {
    int fl = lit + NVARS; if (fl >= NLITS) fl -= NLITS;
    v = num[fl * UNITD + (j - UNITD)] / (den[fl] + 1e-9f);
  }
  X[i] = (_Float16)v;
}
// column-wise stats over 20000 rows (lit_norm, axis=0)
__global__ __launch_bounds__(256) void colstats_kernel(const float* __restrict__ x,
                                                       float* __restrict__ mean,
                                                       float* __restrict__ rvar) {
  __shared__ float sA[256], sB[256];
  int col = blockIdx.x;
  float s = 0.f, s2 = 0.f;
  for (int r = threadIdx.x; r < NLITS; r += blockDim.x) {
    float v = x[(size_t)r * FDIM + col]; s += v; s2 += v * v;
  }
  sA[threadIdx.x] = s; sB[threadIdx.x] = s2; __syncthreads();
  for (int off = 128; off > 0; off >>= 1) {
    if (threadIdx.x < off) { sA[threadIdx.x] += sA[threadIdx.x + off]; sB[threadIdx.x] += sB[threadIdx.x + off]; }
    __syncthreads();
  }
  if (threadIdx.x == 0) {
    float mu = sA[0] / (float)NLITS;
    float var = sB[0] / (float)NLITS - mu * mu;
    mean[col] = mu; rvar[col] = rsqrtf(var + 1e-6f);
  }
}
__global__ void litnorm_kernel(const float* __restrict__ raw, const float* __restrict__ mean,
                               const float* __restrict__ rvar, float* __restrict__ lout) {
  int i = blockIdx.x * blockDim.x + threadIdx.x;
  if (i >= NLITS * FDIM) return;
  int c = i & (FDIM - 1);
  lout[i] = (raw[i] - mean[c]) * rvar[c];
}
__global__ void pack_out_kernel(const float* __restrict__ lout, _Float16* __restrict__ X) {
  int i = blockIdx.x * blockDim.x + threadIdx.x;
  if (i >= NVARS * 256) return;
  int v = i / 256, c = i - v * 256;
  float val = (c < 128) ? lout[v * FDIM + c] : lout[(v + NVARS) * FDIM + (c - 128)];
  X[i] = (_Float16)val;
}
// one wave per variable: logits = 0.25 * h2 . out_w3
__global__ void logits_kernel(const _Float16* __restrict__ h, const float* __restrict__ w3,
                              float* __restrict__ out) {
  int v = blockIdx.x * (blockDim.x >> 5) + (threadIdx.x >> 5);
  int lane = threadIdx.x & 31;
  if (v >= NVARS) return;
  float s = 0.f;
  for (int k = lane; k < FDIM; k += 32) s += (float)h[v * FDIM + k] * w3[k];
  for (int off = 16; off > 0; off >>= 1) s += __shfl_down(s, off, 32);
  if (lane == 0) out[v] = 0.25f * s;
}
__global__ void loss_kernel(const float* __restrict__ logits, const int* __restrict__ cvar,
                            const int* __restrict__ csign, float* __restrict__ acc) {
  int c = blockIdx.x * blockDim.x + threadIdx.x;
  if (c >= NCLS) return;
  int e0 = 3 * c; float s = 0.f;
#pragma unroll
  for (int t = 0; t < 3; ++t) {
    float sg = 1.f - 2.f * (float)csign[e0 + t];
    s += softplusf(logits[cvar[e0 + t]] * sg);
  }
  float vc = __expf(-s);
  atomicAdd(acc, -logf(1.f - vc + 1e-8f));
}
__global__ void finalize_kernel(const float* __restrict__ acc, float* __restrict__ out) {
  if (blockIdx.x == 0 && threadIdx.x == 0)
    out[0] = 0.25f * (acc[0] + acc[1] + acc[2] + acc[3]);
}

// ============================================================
extern "C" void kernel_launch(void* const* d_in, const int* in_sizes, int n_in,
                              void* d_out, int out_size, void* d_ws, size_t ws_size,
                              hipStream_t stream) {
  (void)in_sizes; (void)n_in; (void)out_size; (void)ws_size;
  const float* vq_w1  = (const float*)d_in[0];   // [264,128]
  const float* vq_w2  = (const float*)d_in[1];   // [128,128]
  const float* vq_w3  = (const float*)d_in[2];   // [128,64]
  const float* att_wq = (const float*)d_in[3];   // [224,128]
  const float* att_wm = (const float*)d_in[4];   // [224,128]
  const float* att_v  = (const float*)d_in[5];   // [128]
  const float* lm_w1  = (const float*)d_in[6];   // [448,128]
  const float* lm_w2  = (const float*)d_in[7];   // [128,128]
  const float* lm_w3  = (const float*)d_in[8];   // [128,128]
  const float* out_w1 = (const float*)d_in[9];   // [256,128]
  const float* out_w2 = (const float*)d_in[10];  // [128,128]
  const float* out_w3 = (const float*)d_in[11];  // [128,1]
  const float* noise  = (const float*)d_in[12];  // [4,20000,4]
  const int*   cvar   = (const int*)d_in[13];    // [126000]
  const int*   csign  = (const int*)d_in[14];    // [126000]
  float* out = (float*)d_out;                    // [10000 logits | 1 loss]

  // ---- bump allocator on workspace ----
  char* base = (char*)d_ws; size_t off = 0;
  auto alloc = [&](size_t bytes) -> void* {
    void* r = base + off; off += (bytes + 255) & ~(size_t)255; return r;
  };
  float*     lout  = (float*)alloc((size_t)NLITS * FDIM * 4);
  _Float16*  hA    = (_Float16*)alloc((size_t)NLITS * FDIM * 2);
  _Float16*  hB    = (_Float16*)alloc((size_t)NLITS * FDIM * 2);
  _Float16*  Xvq   = (_Float16*)alloc((size_t)NVARS * 288 * 2);  // reused as Xout
  float*     query = (float*)alloc((size_t)NVARS * QD * 4);
  float*     clls  = (float*)alloc((size_t)NCLS * QD * 4);
  float*     lg    = (float*)alloc((size_t)NVARS * QD * 4);
  float*     llo   = (float*)alloc((size_t)NLITS * QD * 4);
  float*     lituF = (float*)alloc((size_t)NLITS * UNITD * 4);
  _Float16*  lituH = (_Float16*)alloc((size_t)NLITS * UNITD * 2);
  float*     clsuF = (float*)alloc((size_t)NCLS * UNITD * 4);
  _Float16*  clsuH = (_Float16*)alloc((size_t)NCLS * UNITD * 2); // reused as Xlm
  float*     qp    = (float*)alloc((size_t)NLITS * FDIM * 4);
  float*     mp    = (float*)alloc((size_t)NCLS * FDIM * 4);     // reused as lraw
  float*     score = (float*)alloc((size_t)EDG * 4);
  unsigned*  smax  = (unsigned*)alloc((size_t)NLITS * 4);
  float*     den   = (float*)alloc((size_t)NLITS * 4);
  float*     num   = (float*)alloc((size_t)NLITS * UNITD * 4);
  float*     mean  = (float*)alloc(FDIM * 4);
  float*     rvar  = (float*)alloc(FDIM * 4);
  float*     lacc  = (float*)alloc(ROUNDS * 4);
  // f16 transposed weights
  _Float16* vq1t = (_Float16*)alloc((size_t)128 * 288 * 2);
  _Float16* vq2t = (_Float16*)alloc((size_t)128 * 128 * 2);
  _Float16* vq3t = (_Float16*)alloc((size_t)64  * 128 * 2);
  _Float16* awqt = (_Float16*)alloc((size_t)128 * 224 * 2);
  _Float16* awmt = (_Float16*)alloc((size_t)128 * 224 * 2);
  _Float16* lm1t = (_Float16*)alloc((size_t)128 * 448 * 2);
  _Float16* lm2t = (_Float16*)alloc((size_t)128 * 128 * 2);
  _Float16* lm3t = (_Float16*)alloc((size_t)128 * 128 * 2);
  _Float16* o1t  = (_Float16*)alloc((size_t)128 * 256 * 2);
  _Float16* o2t  = (_Float16*)alloc((size_t)128 * 128 * 2);
  // safe aliases (lifetimes disjoint)
  _Float16* Xlm  = clsuH;  // written after att GEMMs consume clsuH
  _Float16* Xout = Xvq;    // written after vq GEMMs consume Xvq
  float*    lraw = mp;     // written after attention consumes mp

  auto nb = [](long long n) { return (unsigned)((n + 255) / 256); };

  // ---- weight conversion (transpose + pad, f32 -> f16) ----
  struct WI { const float* s; _Float16* d; int K, N, Kp; };
  const WI wl[10] = {
    {vq_w1, vq1t, 264, 128, 288}, {vq_w2, vq2t, 128, 128, 128},
    {vq_w3, vq3t, 128,  64, 128}, {att_wq, awqt, 224, 128, 224},
    {att_wm, awmt, 224, 128, 224}, {lm_w1, lm1t, 448, 128, 448},
    {lm_w2, lm2t, 128, 128, 128}, {lm_w3, lm3t, 128, 128, 128},
    {out_w1, o1t, 256, 128, 256}, {out_w2, o2t, 128, 128, 128},
  };
  for (int i = 0; i < 10; ++i)
    wcvt_kernel<<<nb((long long)wl[i].N * wl[i].Kp), 256, 0, stream>>>(
        wl[i].s, wl[i].d, wl[i].K, wl[i].N, wl[i].Kp);

  zero_f32_kernel<<<1, 256, 0, stream>>>(lacc, ROUNDS);
  init_lout_kernel<<<nb(NLITS * FDIM), 256, 0, stream>>>(lout);

  for (int step = 0; step < ROUNDS; ++step) {
    // per-round scratch zeroing
    zero_f32_kernel<<<nb(NVARS * QD), 256, 0, stream>>>(lg, NVARS * QD);
    zero_f32_kernel<<<nb(NLITS * QD), 256, 0, stream>>>(llo, NLITS * QD);
    zero_u32_kernel<<<nb(NLITS), 256, 0, stream>>>(smax, NLITS);
    zero_f32_kernel<<<nb(NLITS), 256, 0, stream>>>(den, NLITS);
    zero_f32_kernel<<<nb(NLITS * UNITD), 256, 0, stream>>>(num, NLITS * UNITD);

    // variables_query MLP
    pack_vq_kernel<<<nb(NVARS * 288), 256, 0, stream>>>(lout, noise + (size_t)step * NLITS * 4, Xvq);
    gemm_wmma_kernel<<<NVARS / 16, 256, 0, stream>>>(Xvq, vq1t, nullptr, hA, 288, 128, 1);
    gemm_wmma_kernel<<<NVARS / 16, 256, 0, stream>>>(hA, vq2t, nullptr, hB, 128, 128, 1);
    gemm_wmma_kernel<<<NVARS / 16, 128, 0, stream>>>(hB, vq3t, query, nullptr, 128, 64, 0);

    // clause values + analytic inner grad + literal loss scatter
    clause_kernel<<<nb(NCLS * QD), 256, 0, stream>>>(query, cvar, csign, clls, lg, llo);
    litunit_kernel<<<nb(NLITS * UNITD), 256, 0, stream>>>(lout, lg, llo, lituF, lituH);
    clsunit_kernel<<<nb(NCLS * UNITD), 256, 0, stream>>>(lout, lg, clls, cvar, csign, clsuF, clsuH);

    // attention projections + edge softmax + aggregation
    gemm_wmma_kernel<<<NLITS / 16, 256, 0, stream>>>(lituH, awqt, qp, nullptr, 224, 128, 0);
    gemm_wmma_kernel<<<NCLS / 16, 256, 0, stream>>>(clsuH, awmt, mp, nullptr, 224, 128, 0);
    score_kernel<<<(EDG + 7) / 8, 256, 0, stream>>>(qp, mp, att_v, cvar, csign, score, smax);
    att_acc_kernel<<<nb((long long)EDG * UNITD), 256, 0, stream>>>(score, smax, cvar, csign, clsuF, num, den);

    // literals MLP
    pack_lm_kernel<<<nb(NLITS * 448), 256, 0, stream>>>(lituF, num, den, Xlm);
    gemm_wmma_kernel<<<NLITS / 16, 256, 0, stream>>>(Xlm, lm1t, nullptr, hA, 448, 128, 1);
    gemm_wmma_kernel<<<NLITS / 16, 256, 0, stream>>>(hA, lm2t, nullptr, hB, 128, 128, 1);
    gemm_wmma_kernel<<<NLITS / 16, 256, 0, stream>>>(hB, lm3t, lraw, nullptr, 128, 128, 0);

    // lit_norm over axis 0
    colstats_kernel<<<FDIM, 256, 0, stream>>>(lraw, mean, rvar);
    litnorm_kernel<<<nb(NLITS * FDIM), 256, 0, stream>>>(lraw, mean, rvar, lout);

    // output MLP -> logits
    pack_out_kernel<<<nb(NVARS * 256), 256, 0, stream>>>(lout, Xout);
    gemm_wmma_kernel<<<NVARS / 16, 256, 0, stream>>>(Xout, o1t, nullptr, hA, 256, 128, 1);
    gemm_wmma_kernel<<<NVARS / 16, 256, 0, stream>>>(hA, o2t, nullptr, hB, 128, 128, 1);
    logits_kernel<<<(NVARS + 7) / 8, 256, 0, stream>>>(hB, out_w3, out);

    // step loss
    loss_kernel<<<nb(NCLS), 256, 0, stream>>>(out, cvar, csign, lacc + step);
  }
  finalize_kernel<<<1, 32, 0, stream>>>(lacc, out + NVARS);
}
